// TransitionNER_88673894793866
// MI455X (gfx1250) — matrix operand order repaired and checked by reading, not compiled, not run on hardware
//
#include <hip/hip_runtime.h>
#include <stdint.h>
#include <stddef.h>

#define HID  1024
#define EMB  1024
#define G4   4096
#define AEMB 128
#define NACT 6

typedef __attribute__((ext_vector_type(16))) __bf16 v16bf;
typedef __attribute__((ext_vector_type(8)))  float  v8f;

// round-to-nearest-away bf16 conversion: single VALU add + high-half move
__device__ __forceinline__ __bf16 f2bf(float f) {
  union { float f; uint32_t u; } v; v.f = f;
  unsigned short s = (unsigned short)((v.u + 0x8000u) >> 16);
  return __builtin_bit_cast(__bf16, s);
}

__device__ __forceinline__ float sigm(float x) { return 1.0f / (1.0f + __expf(-x)); }

// ---------------------------------------------------------------------------
// Deterministic stand-in for the jax xavier-random initial (h,c) states.
// ---------------------------------------------------------------------------
__global__ void k_init(float* inits) {
  int i = blockIdx.x * blockDim.x + threadIdx.x;
  if (i >= 6 * 2 * HID) return;
  uint32_t x = (uint32_t)i * 2654435761u + 0x9E3779B9u;
  x ^= x >> 16; x *= 0x7feb352du; x ^= x >> 15; x *= 0x846ca68bu; x ^= x >> 16;
  float r = (float)(x & 0xFFFFFFu) * (1.0f / 8388608.0f) - 1.0f;
  inits[i] = 0.0765f * r;   // ~sqrt(6/(1+1024))
}

// ---------------------------------------------------------------------------
// Gathers
// ---------------------------------------------------------------------------
__global__ void k_gather_tokens(const int* __restrict__ sentence,
                                const float* __restrict__ word_emb,
                                float* __restrict__ WE) {
  int t = blockIdx.x, j = threadIdx.x;
  WE[(size_t)t * EMB + j] = word_emb[(size_t)sentence[t] * EMB + j];
}

__global__ void k_gather_act(const int* __restrict__ actions,
                             const float* __restrict__ action_emb,
                             float* __restrict__ AE) {
  int t = blockIdx.x, j = threadIdx.x;
  AE[(size_t)t * AEMB + j] = action_emb[(size_t)actions[t] * AEMB + j];
}

// ---------------------------------------------------------------------------
// Sequential bookkeeping: simulate buffer/stack/output/action list lengths and
// derive, per action step, the feature sources, valid-action set and gold pos.
// ---------------------------------------------------------------------------
__global__ void k_book(const int* __restrict__ actions, int A, int S,
                       int* featSrc, int* nvalid, int* goldPos, int* validList,
                       int* redTok, int* redRel) {
  if (threadIdx.x != 0 || blockIdx.x != 0) return;
  for (int r = 0; r < S; ++r) { redTok[r] = 0; redRel[r] = 0; }
  int pops = 0, sp = 0, outPush = 0, nred = 0;
  int stack[512];
  for (int t = 0; t < A; ++t) {
    int gold = actions[t];
    int bufLeft = S - pops;
    int nv = 0, vl[5];
    if (bufLeft > 0) vl[nv++] = 0;                       // SHIFT
    if (sp > 0) { vl[nv++] = 2; vl[nv++] = 3; vl[nv++] = 4; vl[nv++] = 5; }
    else vl[nv++] = 1;                                   // OUT
    nvalid[t] = nv;
    int gp = 0;
    for (int q = 0; q < nv; ++q) if (vl[q] == gold) gp = q;
    goldPos[t] = gp;
    for (int q = 0; q < 5; ++q) validList[t * 5 + q] = (q < nv) ? vl[q] : 0;
    // feature sources: 0=empty, 1=Hbuf, 2=Hstk, 3=HoutC, 4=Hact
    featSrc[t * 8 + 0] = (bufLeft > 0) ? 1 : 0; featSrc[t * 8 + 1] = S - pops;
    featSrc[t * 8 + 2] = (sp > 0) ? 2 : 0;      featSrc[t * 8 + 3] = (sp > 0) ? stack[sp - 1] : 0;
    featSrc[t * 8 + 4] = (outPush > 0) ? 3 : 0; featSrc[t * 8 + 5] = outPush;
    featSrc[t * 8 + 6] = (t > 0) ? 4 : 0;       featSrc[t * 8 + 7] = t;
    // apply action
    if (gold == 0)      { stack[sp++] = pops; pops++; }
    else if (gold == 1) { pops++; outPush++; }
    else {
      if (sp > 0 && nred < S) { redTok[nred] = stack[sp - 1]; redRel[nred] = gold; }
      sp = 0; outPush++; nred++;
    }
  }
}

// ---------------------------------------------------------------------------
// Generic GEMM: Y[M,N] = X[M,K] @ W^T (+bias)(tanh?), W is [N,K] row-major f32.
// bf16 WMMA. Wave tile = 32(M) x 16(N): two accumulators share each B fragment.
// K unrolled by 2 (64-wide chunks): 4 WMMAs per loop body. The 32x64 f32 X
// tile is staged to LDS with async-to-LDS B128 loads (2 per thread) and one
// s_wait_asynccnt per chunk.
// ---------------------------------------------------------------------------
__global__ void k_gemm_wmma(const float* __restrict__ X, const float* __restrict__ W,
                            const float* __restrict__ bias, float* __restrict__ Y,
                            int M, int K, int N, int applyTanh) {
  __shared__ float sX[32][64];
  const int wave = threadIdx.x >> 5;
  const int lane = threadIdx.x & 31;
  const int hi   = lane >> 4;      // K-half selector
  const int ml   = lane & 15;      // row (A) / column (B/C)
  const int m0   = blockIdx.y << 5;
  const int n0   = (blockIdx.x << 7) + (wave << 4);
  int ncol = n0 + ml; if (ncol >= N) ncol = N - 1;
  const float* wbase = W + (size_t)ncol * K;

  v8f c0 = {}, c1 = {};
  const int KC = K >> 6;           // 64-wide K chunks (all K are multiples of 64)
  const int srow  = threadIdx.x >> 3;
  const int sseg  = threadIdx.x & 7;
  int mrow = m0 + srow; if (mrow >= M) mrow = M - 1;
  const float* xrow = X + (size_t)mrow * K;

  for (int kc = 0; kc < KC; ++kc) {
    const int kbase = kc << 6;
    {
      const float* src0 = xrow + kbase + (sseg << 2);
      const float* src1 = src0 + 32;
      uint32_t l0 = (uint32_t)(uintptr_t)&sX[srow][sseg << 2];
      uint32_t l1 = l0 + 32 * 4;
      asm volatile("global_load_async_to_lds_b128 %0, %1, off"
                   :: "v"(l0), "v"((uint64_t)(uintptr_t)src0) : "memory");
      asm volatile("global_load_async_to_lds_b128 %0, %1, off"
                   :: "v"(l1), "v"((uint64_t)(uintptr_t)src1) : "memory");
      asm volatile("s_wait_asynccnt 0" ::: "memory");
    }
    __syncthreads();
#pragma unroll
    for (int s = 0; s < 2; ++s) {
      const int koff = s << 5;
      // A fragments: lane<16 holds K 0..7 & 16..23; lane>=16 holds 8..15 & 24..31
      v16bf a0, a1, b;
#pragma unroll
      for (int j = 0; j < 8; ++j) {
        a0[j]     = f2bf(sX[ml][koff + (hi << 3) + j]);
        a0[8 + j] = f2bf(sX[ml][koff + 16 + (hi << 3) + j]);
        a1[j]     = f2bf(sX[16 + ml][koff + (hi << 3) + j]);
        a1[8 + j] = f2bf(sX[16 + ml][koff + 16 + (hi << 3) + j]);
      }
      // B fragment: K 0..7 & 16..23 (lane<16) / 8..15 & 24..31 (lane>=16) of row ncol (shared by both accumulators)
      const float4* w4 = (const float4*)(wbase + kbase + koff + (hi << 3));
      float wf[16];
      *(float4*)&wf[0]  = w4[0];
      *(float4*)&wf[4]  = w4[1];
      *(float4*)&wf[8]  = w4[4];
      *(float4*)&wf[12] = w4[5];
#pragma unroll
      for (int j = 0; j < 16; ++j) b[j] = f2bf(wf[j]);
      c0 = __builtin_amdgcn_wmma_f32_16x16x32_bf16(false, a0, false, b, (short)0, c0,
                                                   false, false);
      c1 = __builtin_amdgcn_wmma_f32_16x16x32_bf16(false, a1, false, b, (short)0, c1,
                                                   false, false);
    }
    __syncthreads();
  }
  float bv = bias ? bias[ncol] : 0.0f;
#pragma unroll
  for (int r = 0; r < 8; ++r) {
    int m = m0 + r + (hi << 3);
    int n = n0 + ml;
    if (n < N) {
      if (m < M) {
        float v = c0[r] + bv;
        if (applyTanh) v = tanhf(v);
        Y[(size_t)m * N + n] = v;
      }
      if (m + 16 < M) {
        float v = c1[r] + bv;
        if (applyTanh) v = tanhf(v);
        Y[(size_t)(m + 16) * N + n] = v;
      }
    }
  }
}

// ---------------------------------------------------------------------------
// Batched single-step LSTM cell from a fixed init state (stack / entf / entb).
// g = Xg[t] + (Whh@h0) + bih + bhh
// ---------------------------------------------------------------------------
__global__ void k_cell_batch(const float* __restrict__ Xg, const float* __restrict__ hc,
                             const float* __restrict__ bih, const float* __restrict__ bhh,
                             const float* __restrict__ c0, float* __restrict__ H, int T) {
  int t = blockIdx.x, j = threadIdx.x;
  const float* xg = Xg + (size_t)t * G4;
  float g0 = xg[j]           + hc[j]           + bih[j]           + bhh[j];
  float g1 = xg[HID + j]     + hc[HID + j]     + bih[HID + j]     + bhh[HID + j];
  float g2 = xg[2 * HID + j] + hc[2 * HID + j] + bih[2 * HID + j] + bhh[2 * HID + j];
  float g3 = xg[3 * HID + j] + hc[3 * HID + j] + bih[3 * HID + j] + bhh[3 * HID + j];
  float c2 = sigm(g1) * c0[j] + sigm(g0) * tanhf(g2);
  H[(size_t)t * HID + j] = sigm(g3) * tanhf(c2);
}

// ---------------------------------------------------------------------------
// Sequential LSTM recurrence (buffer/action/output chains). One 1024-thread
// workgroup; h broadcast via LDS; Whh stays L2-resident across steps.
// ---------------------------------------------------------------------------
__global__ void __launch_bounds__(1024)
k_chain(const float* __restrict__ Xg, const float* __restrict__ Whh,
        const float* __restrict__ bih, const float* __restrict__ bhh,
        const float* __restrict__ h0, const float* __restrict__ c0,
        float* __restrict__ H, int T, int reverseOrder) {
  __shared__ float sh[HID];
  int j = threadIdx.x;
  float h = h0[j], c = c0[j];
  H[j] = h;
  float bi0 = bih[j]           + bhh[j];
  float bi1 = bih[HID + j]     + bhh[HID + j];
  float bi2 = bih[2 * HID + j] + bhh[2 * HID + j];
  float bi3 = bih[3 * HID + j] + bhh[3 * HID + j];
  const float* w0 = Whh + (size_t)j * HID;
  const float* w1 = Whh + (size_t)(HID + j) * HID;
  const float* w2 = Whh + (size_t)(2 * HID + j) * HID;
  const float* w3 = Whh + (size_t)(3 * HID + j) * HID;
  for (int t = 0; t < T; ++t) {
    sh[j] = h;
    __syncthreads();
    int row = reverseOrder ? (T - 1 - t) : t;
    const float* xg = Xg + (size_t)row * G4;
    if (t + 1 < T) {
      int nrow = reverseOrder ? (T - 2 - t) : (t + 1);
      __builtin_prefetch(Xg + (size_t)nrow * G4 + j, 0, 1);
    }
    float g0 = bi0 + xg[j];
    float g1 = bi1 + xg[HID + j];
    float g2 = bi2 + xg[2 * HID + j];
    float g3 = bi3 + xg[3 * HID + j];
    for (int k = 0; k < HID; k += 4) {
      float4 hv = *(const float4*)&sh[k];
      float4 a0 = *(const float4*)&w0[k];
      float4 a1 = *(const float4*)&w1[k];
      float4 a2 = *(const float4*)&w2[k];
      float4 a3 = *(const float4*)&w3[k];
      g0 += a0.x * hv.x + a0.y * hv.y + a0.z * hv.z + a0.w * hv.w;
      g1 += a1.x * hv.x + a1.y * hv.y + a1.z * hv.z + a1.w * hv.w;
      g2 += a2.x * hv.x + a2.y * hv.y + a2.z * hv.z + a2.w * hv.w;
      g3 += a3.x * hv.x + a3.y * hv.y + a3.z * hv.z + a3.w * hv.w;
    }
    c = sigm(g1) * c + sigm(g0) * tanhf(g2);
    h = sigm(g3) * tanhf(c);
    H[(size_t)(t + 1) * HID + j] = h;
    __syncthreads();
  }
}

// ---------------------------------------------------------------------------
// Entity concat [fh | bh | relation_emb] -> Xcat[S, 2176]
// ---------------------------------------------------------------------------
__global__ void k_xcat(const float* __restrict__ FHv, const float* __restrict__ BHv,
                       const float* __restrict__ relation_emb,
                       const int* __restrict__ redTok, const int* __restrict__ redRel,
                       float* __restrict__ Xcat) {
  int r = blockIdx.x, j = threadIdx.x;
  int tok = redTok[r];
  float* row = Xcat + (size_t)r * 2176;
  row[j]       = FHv[(size_t)tok * HID + j];
  row[HID + j] = BHv[(size_t)tok * HID + j];
  if (j < AEMB) row[2 * HID + j] = relation_emb[(size_t)redRel[r] * AEMB + j];
}

// ---------------------------------------------------------------------------
// feats[t] = [emb(buf), emb(stk), emb(out), emb(act)]
// ---------------------------------------------------------------------------
__global__ void k_feats(const int* __restrict__ featSrc, const float* __restrict__ empty_emb,
                        const float* __restrict__ Hbuf, const float* __restrict__ Hstk,
                        const float* __restrict__ HoutC, const float* __restrict__ Hact,
                        float* __restrict__ feats) {
  int t = blockIdx.x, j = threadIdx.x;
#pragma unroll
  for (int s = 0; s < 4; ++s) {
    int src = featSrc[t * 8 + s * 2 + 0];
    int idx = featSrc[t * 8 + s * 2 + 1];
    const float* p =
        (src == 0) ? empty_emb :
        (src == 1) ? Hbuf  + (size_t)idx * HID :
        (src == 2) ? Hstk  + (size_t)idx * HID :
        (src == 3) ? HoutC + (size_t)idx * HID :
                     Hact  + (size_t)idx * HID;
    feats[(size_t)t * G4 + s * HID + j] = p[j];
  }
}

// ---------------------------------------------------------------------------
// logits[t,a] = hsoft[t] . W_act[a] + b_act[a]
// ---------------------------------------------------------------------------
__global__ void k_logits(const float* __restrict__ hsoft, const float* __restrict__ W_act,
                         const float* __restrict__ b_act, float* __restrict__ logits) {
  __shared__ float part[NACT][256];
  int t = blockIdx.x, tid = threadIdx.x;
  float acc[NACT] = {0, 0, 0, 0, 0, 0};
  for (int k = tid; k < HID; k += 256) {
    float x = hsoft[(size_t)t * HID + k];
#pragma unroll
    for (int a = 0; a < NACT; ++a) acc[a] += x * W_act[(size_t)a * HID + k];
  }
#pragma unroll
  for (int a = 0; a < NACT; ++a) part[a][tid] = acc[a];
  __syncthreads();
  for (int s = 128; s > 0; s >>= 1) {
    if (tid < s)
#pragma unroll
      for (int a = 0; a < NACT; ++a) part[a][tid] += part[a][tid + s];
    __syncthreads();
  }
  if (tid < NACT) logits[(size_t)t * NACT + tid] = part[tid][0] + b_act[tid];
}

// ---------------------------------------------------------------------------
// loss = -sum_t logsoftmax(logits[t, valid])[gold]
// ---------------------------------------------------------------------------
__global__ void k_loss(const float* __restrict__ logits, const int* __restrict__ nvalid,
                       const int* __restrict__ goldPos, const int* __restrict__ validList,
                       float* __restrict__ out, int A) {
  if (threadIdx.x != 0 || blockIdx.x != 0) return;
  float acc = 0.0f;
  for (int t = 0; t < A; ++t) {
    int nv = nvalid[t];
    if (nv <= 1) continue;
    float l[5], m = -1e30f;
    for (int q = 0; q < nv; ++q) {
      l[q] = logits[(size_t)t * NACT + validList[t * 5 + q]];
      m = fmaxf(m, l[q]);
    }
    float s = 0.0f;
    for (int q = 0; q < nv; ++q) s += expf(l[q] - m);
    acc += l[goldPos[t]] - m - logf(s);
  }
  out[0] = -acc;
}

// ---------------------------------------------------------------------------
extern "C" void kernel_launch(void* const* d_in, const int* in_sizes, int n_in,
                              void* d_out, int out_size, void* d_ws, size_t ws_size,
                              hipStream_t stream) {
  (void)n_in; (void)out_size; (void)ws_size;
  const int*   sentence     = (const int*)d_in[0];
  const int*   actions      = (const int*)d_in[2];
  const float* word_emb     = (const float*)d_in[3];
  const float* action_emb   = (const float*)d_in[4];
  const float* relation_emb = (const float*)d_in[5];
  const float* empty_emb    = (const float*)d_in[6];
  const float *Wih[6], *Whh[6], *bih[6], *bhh[6];
  // cell order: 0=buffer 1=stack 2=action 3=output 4=entf 5=entb
  for (int c = 0; c < 6; ++c) {
    Wih[c] = (const float*)d_in[7 + 4 * c + 0];
    Whh[c] = (const float*)d_in[7 + 4 * c + 1];
    bih[c] = (const float*)d_in[7 + 4 * c + 2];
    bhh[c] = (const float*)d_in[7 + 4 * c + 3];
  }
  const float* W_soft = (const float*)d_in[31];
  const float* b_soft = (const float*)d_in[32];
  const float* W_act  = (const float*)d_in[33];
  const float* b_act  = (const float*)d_in[34];
  const float* W_ent  = (const float*)d_in[35];
  const float* b_ent  = (const float*)d_in[36];

  const int S = in_sizes[0];   // 128 tokens
  const int A = in_sizes[2];   // 256 actions

  // ----- workspace layout (floats) -----
  float* ws = (float*)d_ws;
  size_t o = 0;
  float* INITS  = ws + o; o += 6 * 2 * HID;        // (h0,c0) x 6 cells
  float* WE     = ws + o; o += (size_t)S * EMB;
  float* AE     = ws + o; o += (size_t)A * AEMB;
  float* XBUF   = ws + o; o += (size_t)S * G4;
  float* XSTK   = ws + o; o += (size_t)S * G4;
  float* XENTF  = ws + o; o += (size_t)S * G4;
  float* XENTB  = ws + o; o += (size_t)S * G4;
  float* XACT   = ws + o; o += (size_t)A * G4;
  float* HCSTK  = ws + o; o += G4;
  float* HCENTF = ws + o; o += G4;
  float* HCENTB = ws + o; o += G4;
  float* HSTK   = ws + o; o += (size_t)S * HID;
  float* FHv    = ws + o; o += (size_t)S * HID;
  float* BHv    = ws + o; o += (size_t)S * HID;
  float* XCAT   = ws + o; o += (size_t)S * 2176;
  float* EIN    = ws + o; o += (size_t)S * HID;
  float* XOUT   = ws + o; o += (size_t)S * G4;
  float* HBUF   = ws + o; o += (size_t)(S + 1) * HID;
  float* HACT   = ws + o; o += (size_t)(A + 1) * HID;
  float* HOUTC  = ws + o; o += (size_t)(S + 1) * HID;
  float* FEATS  = ws + o; o += (size_t)A * G4;
  float* HSOFT  = ws + o; o += (size_t)A * HID;
  float* LOGITS = ws + o; o += (size_t)A * NACT;
  int* ib        = (int*)(ws + o);
  int* featSrc   = ib;                 ib += A * 8;
  int* nvalid    = ib;                 ib += A;
  int* goldPos   = ib;                 ib += A;
  int* validList = ib;                 ib += A * 5;
  int* redTok    = ib;                 ib += S;
  int* redRel    = ib;                 ib += S;

  // ----- setup -----
  k_init<<<(6 * 2 * HID + 255) / 256, 256, 0, stream>>>(INITS);
  k_gather_tokens<<<S, EMB, 0, stream>>>(sentence, word_emb, WE);
  k_gather_act<<<A, AEMB, 0, stream>>>(actions, action_emb, AE);
  k_book<<<1, 1, 0, stream>>>(actions, A, S, featSrc, nvalid, goldPos, validList,
                              redTok, redRel);

  // ----- batched input projections (WMMA GEMMs) -----
  dim3 gS(G4 / 128, (S + 31) / 32), gA(G4 / 128, (A + 31) / 32), g1(G4 / 128, 1);
  k_gemm_wmma<<<gS, 256, 0, stream>>>(WE, Wih[0], nullptr, XBUF, S, EMB, G4, 0);
  k_gemm_wmma<<<gS, 256, 0, stream>>>(WE, Wih[1], nullptr, XSTK, S, EMB, G4, 0);
  k_gemm_wmma<<<gS, 256, 0, stream>>>(WE, Wih[4], nullptr, XENTF, S, EMB, G4, 0);
  k_gemm_wmma<<<gS, 256, 0, stream>>>(WE, Wih[5], nullptr, XENTB, S, EMB, G4, 0);
  k_gemm_wmma<<<gA, 256, 0, stream>>>(AE, Wih[2], nullptr, XACT, A, AEMB, G4, 0);

  // h-side constants for the single-step cells (M=1 GEMMs)
  k_gemm_wmma<<<g1, 256, 0, stream>>>(INITS + 1 * 2 * HID, Whh[1], nullptr, HCSTK, 1, HID, G4, 0);
  k_gemm_wmma<<<g1, 256, 0, stream>>>(INITS + 4 * 2 * HID, Whh[4], nullptr, HCENTF, 1, HID, G4, 0);
  k_gemm_wmma<<<g1, 256, 0, stream>>>(INITS + 5 * 2 * HID, Whh[5], nullptr, HCENTB, 1, HID, G4, 0);

  // ----- batched single-step cells: stack tops, entity fwd/bwd -----
  k_cell_batch<<<S, HID, 0, stream>>>(XSTK, HCSTK, bih[1], bhh[1],
                                      INITS + 1 * 2 * HID + HID, HSTK, S);
  k_cell_batch<<<S, HID, 0, stream>>>(XENTF, HCENTF, bih[4], bhh[4],
                                      INITS + 4 * 2 * HID + HID, FHv, S);
  k_cell_batch<<<S, HID, 0, stream>>>(XENTB, HCENTB, bih[5], bhh[5],
                                      INITS + 5 * 2 * HID + HID, BHv, S);

  // ----- entity projection and output-chain inputs -----
  k_xcat<<<S, HID, 0, stream>>>(FHv, BHv, relation_emb, redTok, redRel, XCAT);
  k_gemm_wmma<<<dim3(EMB / 128, (S + 31) / 32), 256, 0, stream>>>(
      XCAT, W_ent, b_ent, EIN, S, 2176, EMB, 0);
  k_gemm_wmma<<<gS, 256, 0, stream>>>(EIN, Wih[3], nullptr, XOUT, S, EMB, G4, 0);

  // ----- the three true sequential recurrences -----
  k_chain<<<1, HID, 0, stream>>>(XBUF, Whh[0], bih[0], bhh[0],
                                 INITS + 0, INITS + HID, HBUF, S, 1);
  k_chain<<<1, HID, 0, stream>>>(XACT, Whh[2], bih[2], bhh[2],
                                 INITS + 2 * 2 * HID, INITS + 2 * 2 * HID + HID, HACT, A, 0);
  k_chain<<<1, HID, 0, stream>>>(XOUT, Whh[3], bih[3], bhh[3],
                                 INITS + 3 * 2 * HID, INITS + 3 * 2 * HID + HID, HOUTC, S, 0);

  // ----- softmax head -----
  k_feats<<<A, HID, 0, stream>>>(featSrc, empty_emb, HBUF, HSTK, HOUTC, HACT, FEATS);
  k_gemm_wmma<<<dim3(HID / 128, (A + 31) / 32), 256, 0, stream>>>(
      FEATS, W_soft, b_soft, HSOFT, A, G4, HID, 1);
  k_logits<<<A, 256, 0, stream>>>(HSOFT, W_act, b_act, LOGITS);
  k_loss<<<1, 1, 0, stream>>>(LOGITS, nvalid, goldPos, validList, (float*)d_out, A);
}
